// StackedUnidirectionalEncoder_49950469653290
// MI455X (gfx1250) — matrix operand order repaired and checked by reading, not compiled, not run on hardware
//
#include <hip/hip_runtime.h>
#include <hip/hip_bf16.h>

#define VOCAB 32000
#define EMB   512
#define HID   1024
#define BATCH 64
#define SEQ   512
#define G4H   (4 * HID)

typedef __attribute__((ext_vector_type(16))) __bf16 v16bf;
typedef __attribute__((ext_vector_type(8)))  __bf16 v8bf;
typedef __attribute__((ext_vector_type(8)))  float  v8f;

static __device__ __forceinline__ unsigned short f2bf_bits(float f) {
  union { float f; unsigned u; } c; c.f = f;
  unsigned u = c.u;
  u += 0x7fffu + ((u >> 16) & 1u);   // round-to-nearest-even
  return (unsigned short)(u >> 16);
}

static __device__ __forceinline__ float fast_sigmoid(float x) {
  return 1.0f / (1.0f + __expf(-x));
}
static __device__ __forceinline__ float fast_tanh(float x) {
  return 2.0f / (1.0f + __expf(-2.0f * x)) - 1.0f;
}

// ---- CDNA5 async global->LDS copy (ASYNCcnt path, CDNA5 ISA) ----
// Moves 16 bytes per lane: LDS[lds_byte_off] = MEM[gaddr].
static __device__ __forceinline__ void async_copy_b128(unsigned lds_byte_off, const void* gaddr) {
  asm volatile("global_load_async_to_lds_b128 %0, %1, off"
               :: "v"(lds_byte_off), "v"(gaddr)
               : "memory");
}
static __device__ __forceinline__ void async_wait_all() {
  asm volatile("s_wait_asynccnt 0x0" ::: "memory");
}
// Workgroup-relative LDS byte offset = low 32 bits of the flat address (ISA §10.2).
static __device__ __forceinline__ unsigned lds_offset(const void* p) {
  return (unsigned)(unsigned long long)p;
}

// A tile: 16x32 bf16, row-major source, stride ld (elements).
// ISA layout: lane m=lane&15; lanes0-15 hold K 0..7 (v0-3) and 16..23 (v4-7);
// lanes16-31 hold K 8..15 and 24..31.
template <typename PT>
static __device__ __forceinline__ v16bf load_a_tile(PT p, int ld, int lane) {
  int m  = lane & 15;
  int k0 = (lane >> 4) << 3;            // 0 or 8
  auto r = p + (size_t)m * ld + k0;
  v8bf lo = *(const v8bf*)(&r[0]);      // K k0..k0+7
  v8bf hi = *(const v8bf*)(&r[16]);     // K k0+16..k0+23
  v16bf out;
#pragma unroll
  for (int i = 0; i < 8; ++i) { out[i] = lo[i]; out[i + 8] = hi[i]; }
  return out;
}

// B tile: 32x16 bf16 from N-major (pre-transposed) storage WT[n][k], stride ld.
// ISA layout: lane holds column n=lane&15; lanes0-15 K 0..7,16..23; lanes16-31 K 8..15,24..31.
static __device__ __forceinline__ v16bf load_b_tile(const unsigned short* p, int ld, int lane) {
  int n  = lane & 15;
  int k0 = (lane >> 4) << 3;            // 0 or 8
  const unsigned short* r = p + (size_t)n * ld + k0;
  v8bf lo = *(const v8bf*)(r);
  v8bf hi = *(const v8bf*)(r + 16);
  v16bf out;
#pragma unroll
  for (int i = 0; i < 8; ++i) { out[i] = lo[i]; out[i + 8] = hi[i]; }
  return out;
}

// One LSTM cell step: z = X @ WxT' + Hprev @ WhT' + b ; gates ; c,h update.
// Grid: 32 blocks x 256 threads (8 waves). wave-task = (mtile 0..3, jtile 0..63);
// all 8 waves of a block share one mtile, so the A panels (16 rows x full K,
// contiguous in global) are staged once into LDS with async copies and the
// WMMA A operand is then fed from LDS (ds_load_b128).
__global__ __launch_bounds__(256)
void lstm_cell_wmma(const unsigned short* __restrict__ X,     // [BATCH][Kx] bf16
                    const unsigned short* __restrict__ WxT,   // [4H][Kx]   bf16
                    int Kx,
                    const unsigned short* __restrict__ Hprev, // [BATCH][HID] bf16
                    const unsigned short* __restrict__ WhT,   // [4H][HID]  bf16
                    const float* __restrict__ bias,           // [4H]
                    float* __restrict__ c_state,              // [BATCH][HID] in/out
                    float* __restrict__ h_state,              // [BATCH][HID] out f32
                    unsigned short* __restrict__ h_bf16)      // [BATCH][HID] out bf16
{
  __shared__ __attribute__((aligned(16))) unsigned short sX[16 * HID]; // x panel (Kx<=1024)
  __shared__ __attribute__((aligned(16))) unsigned short sH[16 * HID]; // h panel

  const int tidb = threadIdx.x;
  const int lane = tidb & 31;
  const int wave = tidb >> 5;
  const int tid  = blockIdx.x * 8 + wave;  // 0..255
  const int jt   = tid & 63;
  const int mt   = tid >> 6;               // same for all waves of a block
  const int m0   = mt * 16;
  const int j0   = jt * 16;

  // ---- stage A panels: async global -> LDS, then wait + barrier ----
  {
    const char* gx = (const char*)(X + (size_t)m0 * Kx);
    const unsigned sx0 = lds_offset(sX);
    const int nbx = 16 * Kx * 2;
    for (int ofs = tidb * 16; ofs < nbx; ofs += 256 * 16)
      async_copy_b128(sx0 + (unsigned)ofs, gx + ofs);

    const char* gh = (const char*)(Hprev + (size_t)m0 * HID);
    const unsigned sh0 = lds_offset(sH);
    const int nbh = 16 * HID * 2;
    for (int ofs = tidb * 16; ofs < nbh; ofs += 256 * 16)
      async_copy_b128(sh0 + (unsigned)ofs, gh + ofs);

    async_wait_all();      // my wave's async copies landed in LDS
    __syncthreads();       // everyone's copies visible
  }

  v8f acc[4] = {};

  // x-part: K = Kx (A from LDS, B from global/L2)
  for (int k = 0; k < Kx; k += 32) {
    v16bf a = load_a_tile((const unsigned short*)sX + k, Kx, lane);
#pragma unroll
    for (int g = 0; g < 4; ++g) {
      v16bf b = load_b_tile(WxT + (size_t)(g * HID + j0) * Kx + k, Kx, lane);
      acc[g] = __builtin_amdgcn_wmma_f32_16x16x32_bf16(
          false, a, false, b, (short)0, acc[g], false, false);
    }
  }
  // h-part: K = HID
  for (int k = 0; k < HID; k += 32) {
    v16bf a = load_a_tile((const unsigned short*)sH + k, HID, lane);
#pragma unroll
    for (int g = 0; g < 4; ++g) {
      v16bf b = load_b_tile(WhT + (size_t)(g * HID + j0) * HID + k, HID, lane);
      acc[g] = __builtin_amdgcn_wmma_f32_16x16x32_bf16(
          false, a, false, b, (short)0, acc[g], false, false);
    }
  }

  // Epilogue: C/D layout — vgpr v, lanes0-15 -> row v, lanes16-31 -> row v+8; col = lane&15.
  const int n    = lane & 15;
  const int rofs = (lane >> 4) * 8;
  const int col  = j0 + n;
  const float bi = bias[0 * HID + col];
  const float bf = bias[1 * HID + col];
  const float bo = bias[2 * HID + col];
  const float bg = bias[3 * HID + col];
#pragma unroll
  for (int v = 0; v < 8; ++v) {
    const int m = m0 + rofs + v;
    const float ig = fast_sigmoid(acc[0][v] + bi);
    const float fg = fast_sigmoid(acc[1][v] + bf);
    const float og = fast_sigmoid(acc[2][v] + bo);
    const float gt = fast_tanh(acc[3][v] + bg);
    const size_t idx = (size_t)m * HID + col;
    const float cn = fg * c_state[idx] + ig * gt;
    const float hn = og * fast_tanh(cn);
    c_state[idx] = cn;
    h_state[idx] = hn;
    h_bf16[idx]  = f2bf_bits(hn);
  }
}

// Row LayerNorm over HID=1024; one block (256 threads) per batch row.
__global__ __launch_bounds__(256)
void layernorm_row(const float* __restrict__ h,      // [BATCH][HID]
                   const float* __restrict__ gamma,  // [HID]
                   const float* __restrict__ beta,   // [HID]
                   unsigned short* __restrict__ out_bf16, // nullable [BATCH][HID]
                   float* __restrict__ out_f32,           // nullable, row stride below
                   size_t out_f32_row_stride)
{
  __shared__ float s_sum[8], s_ssq[8];
  const int b = blockIdx.x;
  const float* row = h + (size_t)b * HID;
  float vals[4];
  float s = 0.f, ss = 0.f;
#pragma unroll
  for (int i = 0; i < 4; ++i) {
    const float x = row[threadIdx.x + 256 * i];
    vals[i] = x; s += x; ss += x * x;
  }
#pragma unroll
  for (int off = 16; off > 0; off >>= 1) {
    s  += __shfl_xor(s,  off, 32);
    ss += __shfl_xor(ss, off, 32);
  }
  const int wave = threadIdx.x >> 5, lane = threadIdx.x & 31;
  if (lane == 0) { s_sum[wave] = s; s_ssq[wave] = ss; }
  __syncthreads();
  if (wave == 0) {
    float a = (lane < 8) ? s_sum[lane] : 0.f;
    float q = (lane < 8) ? s_ssq[lane] : 0.f;
#pragma unroll
    for (int off = 4; off > 0; off >>= 1) {
      a += __shfl_xor(a, off, 32);
      q += __shfl_xor(q, off, 32);
    }
    if (lane == 0) { s_sum[0] = a; s_ssq[0] = q; }
  }
  __syncthreads();
  const float mean = s_sum[0] * (1.0f / HID);
  const float var  = s_ssq[0] * (1.0f / HID) - mean * mean;
  const float rstd = rsqrtf(var + 1e-5f);
#pragma unroll
  for (int i = 0; i < 4; ++i) {
    const int c = threadIdx.x + 256 * i;
    const float y = (vals[i] - mean) * rstd * gamma[c] + beta[c];
    if (out_bf16) out_bf16[(size_t)b * HID + c] = f2bf_bits(y);
    if (out_f32)  out_f32[(size_t)b * out_f32_row_stride + c] = y;
  }
}

// Gather embedding rows -> bf16 [SEQ][BATCH][EMB] (timestep-contiguous)
__global__ __launch_bounds__(256)
void gather_embed(const int* __restrict__ src,    // [BATCH][SEQ]
                  const float* __restrict__ emb,  // [VOCAB][EMB]
                  unsigned short* __restrict__ X) // [SEQ][BATCH][EMB]
{
  const int t = blockIdx.x / BATCH;
  const int b = blockIdx.x % BATCH;
  const int tok = src[(size_t)b * SEQ + t];
  const float* e = emb + (size_t)tok * EMB;
  unsigned short* o = X + ((size_t)t * BATCH + b) * EMB;
  for (int i = threadIdx.x; i < EMB; i += 256) o[i] = f2bf_bits(e[i]);
}

// fp32 [K][4H] -> bf16 N-major [4H][K]
__global__ __launch_bounds__(256)
void transpose_w_bf16(const float* __restrict__ W,    // [K][4H]
                      unsigned short* __restrict__ WT,// [4H][K]
                      int K)
{
  const size_t idx = (size_t)blockIdx.x * 256 + threadIdx.x;
  const size_t total = (size_t)K * G4H;
  if (idx >= total) return;
  const size_t n = idx / K;
  const size_t k = idx % K;
  WT[idx] = f2bf_bits(W[k * G4H + n]);
}

__global__ __launch_bounds__(256)
void zero_bytes16(uint4* __restrict__ p, size_t n16)
{
  const size_t i = (size_t)blockIdx.x * 256 + threadIdx.x;
  if (i < n16) p[i] = make_uint4(0u, 0u, 0u, 0u);
}

// final_hidden = concat(h1,h2) [B,2H]; final_cell = concat(c1,c2) [B,2H]
__global__ __launch_bounds__(256)
void write_final_states(const float* __restrict__ h1, const float* __restrict__ h2,
                        const float* __restrict__ c1, const float* __restrict__ c2,
                        float* __restrict__ out_tail)
{
  const int idx = blockIdx.x * 256 + threadIdx.x;  // over BATCH*2*HID
  if (idx >= BATCH * 2 * HID) return;
  const int b = idx / (2 * HID);
  const int c = idx % (2 * HID);
  const float hv = (c < HID) ? h1[(size_t)b * HID + c] : h2[(size_t)b * HID + (c - HID)];
  const float cv = (c < HID) ? c1[(size_t)b * HID + c] : c2[(size_t)b * HID + (c - HID)];
  out_tail[idx] = hv;
  out_tail[BATCH * 2 * HID + idx] = cv;
}

extern "C" void kernel_launch(void* const* d_in, const int* in_sizes, int n_in,
                              void* d_out, int out_size, void* d_ws, size_t ws_size,
                              hipStream_t stream)
{
  (void)in_sizes; (void)n_in; (void)out_size; (void)ws_size;
  const int*   src = (const int*)  d_in[0];
  const float* emb = (const float*)d_in[1];
  const float* Wx1 = (const float*)d_in[2];
  const float* Wh1 = (const float*)d_in[3];
  const float* b1  = (const float*)d_in[4];
  const float* Wx2 = (const float*)d_in[5];
  const float* Wh2 = (const float*)d_in[6];
  const float* b2  = (const float*)d_in[7];
  const float* g1  = (const float*)d_in[8];
  const float* bl1 = (const float*)d_in[9];
  const float* g2  = (const float*)d_in[10];
  const float* bl2 = (const float*)d_in[11];
  float* out = (float*)d_out;

  char* ws = (char*)d_ws;
  size_t off = 0;
  auto alloc = [&](size_t bytes) -> char* {
    off = (off + 255) & ~(size_t)255;
    char* p = ws + off;
    off += bytes;
    return p;
  };

  unsigned short* Xb   = (unsigned short*)alloc((size_t)SEQ * BATCH * EMB * 2);
  unsigned short* WxT1 = (unsigned short*)alloc((size_t)G4H * EMB * 2);
  unsigned short* WhT1 = (unsigned short*)alloc((size_t)G4H * HID * 2);
  unsigned short* WxT2 = (unsigned short*)alloc((size_t)G4H * HID * 2);
  unsigned short* WhT2 = (unsigned short*)alloc((size_t)G4H * HID * 2);

  char* state_begin = ws + ((off + 255) & ~(size_t)255);
  float* c1s = (float*)alloc((size_t)BATCH * HID * 4);
  float* c2s = (float*)alloc((size_t)BATCH * HID * 4);
  float* h1s = (float*)alloc((size_t)BATCH * HID * 4);
  float* h2s = (float*)alloc((size_t)BATCH * HID * 4);
  unsigned short* h1b = (unsigned short*)alloc((size_t)2 * BATCH * HID * 2); // double buffer
  unsigned short* h2b = (unsigned short*)alloc((size_t)2 * BATCH * HID * 2);
  unsigned short* x2b = (unsigned short*)alloc((size_t)BATCH * HID * 2);
  char* state_end = ws + off;

  // 1) zero all recurrent state (h/c start at 0; ws is poisoned)
  {
    size_t n16 = (size_t)(state_end - state_begin + 15) / 16;
    zero_bytes16<<<(unsigned)((n16 + 255) / 256), 256, 0, stream>>>((uint4*)state_begin, n16);
  }
  // 2) embedding gather -> bf16, timestep-major
  gather_embed<<<SEQ * BATCH, 256, 0, stream>>>(src, emb, Xb);
  // 3) weight transposes -> bf16 N-major (stays hot in 192MB L2 afterwards)
  {
    const size_t nE = (size_t)G4H * EMB, nH = (size_t)G4H * HID;
    transpose_w_bf16<<<(unsigned)((nE + 255) / 256), 256, 0, stream>>>(Wx1, WxT1, EMB);
    transpose_w_bf16<<<(unsigned)((nH + 255) / 256), 256, 0, stream>>>(Wh1, WhT1, HID);
    transpose_w_bf16<<<(unsigned)((nH + 255) / 256), 256, 0, stream>>>(Wx2, WxT2, HID);
    transpose_w_bf16<<<(unsigned)((nH + 255) / 256), 256, 0, stream>>>(Wh2, WhT2, HID);
  }

  const size_t HB = (size_t)BATCH * HID;
  // 4) sequential recurrence
  for (int t = 0; t < SEQ; ++t) {
    const unsigned short* xt = Xb + (size_t)t * BATCH * EMB;
    unsigned short* h1_rd = h1b + (size_t)(t & 1) * HB;
    unsigned short* h1_wr = h1b + (size_t)((t + 1) & 1) * HB;
    unsigned short* h2_rd = h2b + (size_t)(t & 1) * HB;
    unsigned short* h2_wr = h2b + (size_t)((t + 1) & 1) * HB;

    lstm_cell_wmma<<<32, 256, 0, stream>>>(xt, WxT1, EMB, h1_rd, WhT1, b1,
                                           c1s, h1s, h1_wr);
    layernorm_row<<<BATCH, 256, 0, stream>>>(h1s, g1, bl1, x2b, nullptr, 0);
    lstm_cell_wmma<<<32, 256, 0, stream>>>(x2b, WxT2, HID, h2_rd, WhT2, b2,
                                           c2s, h2s, h2_wr);
    layernorm_row<<<BATCH, 256, 0, stream>>>(h2s, g2, bl2, nullptr,
                                             out + (size_t)t * HID,
                                             (size_t)SEQ * HID);
  }
  // 5) final hidden/cell concat
  write_final_states<<<(BATCH * 2 * HID + 255) / 256, 256, 0, stream>>>(
      h1s, h2s, c1s, c2s, out + (size_t)BATCH * SEQ * HID);
}
